// XLMRCorefModel_73572789780979
// MI455X (gfx1250) — matrix operand order repaired and checked by reading, not compiled, not run on hardware
//
#include <hip/hip_runtime.h>
#include <hip/hip_bf16.h>
#include <stdint.h>

// ---------------------------------------------------------------------------
// Problem constants (B=1)
// ---------------------------------------------------------------------------
#define S     512
#define H     768
#define E     1024
#define L     50
#define SPANW 10
#define NSPAN 5075          // 503*10 + 45
#define NPAD  5120          // NSPAN rounded up to 64 (block m-tile)
#define K3H   2304          // 3*H
#define KSTEPS (K3H / 32)   // 72

typedef __attribute__((ext_vector_type(16))) __bf16 v16bf;
typedef __attribute__((ext_vector_type(8)))  float  v8f;
typedef __attribute__((ext_vector_type(4)))  int    v4i;

union FragB16 {
    v16bf v;
    uint4 q[2];
};

// ---------------------------------------------------------------------------
// CDNA5 async global->LDS copy (ASYNCcnt) with synchronous fallback.
// Probe-discovered signature: (v4i addrspace(1)*, v4i addrspace(3)*, Ii, Ii)
// ---------------------------------------------------------------------------
#if __has_builtin(__builtin_amdgcn_global_load_async_to_lds_b128)
#define ASYNC_LDS 1
#else
#define ASYNC_LDS 0
#endif

typedef __attribute__((address_space(1))) v4i gv4i;
typedef __attribute__((address_space(3))) v4i lv4i;

__device__ __forceinline__ void stage_b128(const __bf16* g, __bf16* l) {
#if ASYNC_LDS
    __builtin_amdgcn_global_load_async_to_lds_b128(
        (gv4i*)(v4i*)g, (lv4i*)(v4i*)l, 0, 0);
#else
    *reinterpret_cast<uint4*>(l) = *reinterpret_cast<const uint4*>(g);
#endif
}

__device__ __forceinline__ void wait_async_le1() {
#if ASYNC_LDS
#if __has_builtin(__builtin_amdgcn_s_wait_asynccnt)
    __builtin_amdgcn_s_wait_asynccnt(1);
#else
    asm volatile("s_wait_asynccnt 0x1" ::: "memory");
#endif
#endif
}

__device__ __forceinline__ void wait_async_le0() {
#if ASYNC_LDS
#if __has_builtin(__builtin_amdgcn_s_wait_asynccnt)
    __builtin_amdgcn_s_wait_asynccnt(0);
#else
    asm volatile("s_wait_asynccnt 0x0" ::: "memory");
#endif
#endif
}

// ---------------------------------------------------------------------------
// Workspace layout (bytes), 256B aligned sections
//   gcn_acc : S*H f32            @ 0          (1,572,864)
//   emb     : S*H f32            @ 1,572,864  (1,572,864)
//   tok     : S   f32            @ 3,145,728  (pad to 4096)
//   spanv   : NPAD*K3H bf16      @ 3,149,824  (23,592,960)
//   w1t     : K3H*K3H bf16       @ 26,742,784 (10,616,832)
//   total ~ 37.4 MB
// ---------------------------------------------------------------------------
#define WS_GCN  0
#define WS_EMB  1572864
#define WS_TOK  3145728
#define WS_SPAN 3149824
#define WS_W1T  26742784

// ---------------------------------------------------------------------------
// K0: zero the scatter accumulator, seed output with ms_b2
// ---------------------------------------------------------------------------
__global__ void k_init(float* __restrict__ gcn_acc, float* __restrict__ out,
                       const float* __restrict__ b2) {
    int i = blockIdx.x * 256 + threadIdx.x;
    if (i < S * H) gcn_acc[i] = 0.0f;
    if (i < NSPAN) out[i] = b2[0];
}

// ---------------------------------------------------------------------------
// K1: per-edge matvec  msgs[e] = hidden[src] @ W[label] + b[label],
//     scatter-add into gcn_acc[tgt].  All 50 W matrices (113 MB) fit in the
//     192 MB L2, so the per-edge matvec path (1.2 GFLOP total, 50x less work
//     than the reference's label scan) is L2-resident and cheap.
// ---------------------------------------------------------------------------
__global__ void __launch_bounds__(256) k_gcn_edge(
    const float* __restrict__ hidden, const float* __restrict__ gcnW,
    const float* __restrict__ gcnb,   const int* __restrict__ esrc,
    const int* __restrict__ etgt,     const int* __restrict__ elab,
    float* __restrict__ gcn_acc) {
    __shared__ float xs[H];
    const int e = blockIdx.x;
    const int t = threadIdx.x;
    const int src = esrc[e], tgt = etgt[e], l = elab[e];
    for (int h = t; h < H; h += 256) xs[h] = hidden[src * H + h];
    __syncthreads();

    const float* Wl = gcnW + (size_t)l * H * H;
    float a0 = gcnb[l * H + t];
    float a1 = gcnb[l * H + t + 256];
    float a2 = gcnb[l * H + t + 512];
    for (int h = 0; h < H; ++h) {
        const float xv = xs[h];
        const float* row = Wl + (size_t)h * H;
        a0 = fmaf(xv, row[t],       a0);
        a1 = fmaf(xv, row[t + 256], a1);
        a2 = fmaf(xv, row[t + 512], a2);
    }
    atomicAdd(gcn_acc + (size_t)tgt * H + t,       a0);
    atomicAdd(gcn_acc + (size_t)tgt * H + t + 256, a1);
    atomicAdd(gcn_acc + (size_t)tgt * H + t + 512, a2);
}

// ---------------------------------------------------------------------------
// K2: emb = hidden + relu(gcn_acc); tok[s] = dot(emb[s], attn_w) + attn_b
// ---------------------------------------------------------------------------
__global__ void __launch_bounds__(256) k_emb_tok(
    const float* __restrict__ hidden, const float* __restrict__ gcn_acc,
    const float* __restrict__ attn_w, const float* __restrict__ attn_b,
    float* __restrict__ emb, float* __restrict__ tok) {
    __shared__ float red[256];
    const int s = blockIdx.x, t = threadIdx.x;
    float part = 0.0f;
    for (int h = t; h < H; h += 256) {
        float g = gcn_acc[(size_t)s * H + h];
        g = g > 0.0f ? g : 0.0f;
        const float v = hidden[(size_t)s * H + h] + g;
        emb[(size_t)s * H + h] = v;
        part = fmaf(v, attn_w[h], part);
    }
    red[t] = part;
    __syncthreads();
    for (int o = 128; o > 0; o >>= 1) {
        if (t < o) red[t] += red[t + o];
        __syncthreads();
    }
    if (t == 0) tok[s] = red[0] + attn_b[0];
}

// ---------------------------------------------------------------------------
// K3: build span_vec rows in bf16: [emb[start] | emb[end] | attn_out]
//     Softmax over the <=10 window tokens uses precomputed token scores.
//     Rows n >= NSPAN are zero padding for the WMMA GEMM.
// ---------------------------------------------------------------------------
__global__ void __launch_bounds__(256) k_span(
    const float* __restrict__ emb, const float* __restrict__ tok,
    __bf16* __restrict__ spanv) {
    const int n = blockIdx.x, t = threadIdx.x;
    __bf16* row = spanv + (size_t)n * K3H;
    if (n >= NSPAN) {
        for (int i = t; i < K3H; i += 256) row[i] = (__bf16)0.0f;
        return;
    }
    // decode span index -> (start, end)
    int start, end;
    if (n < 5030) {
        start = n / SPANW;
        end = start + (n % SPANW);
    } else {
        int r = n - 5030;
        start = 503;
        int cnt = 9;
        while (r >= cnt) { r -= cnt; --cnt; ++start; }
        end = start + r;
    }
    const int width = end - start;  // 0..9 inclusive window extent

    float wts[SPANW];
    float mx = -1e30f;
    for (int w = 0; w <= width; ++w) {
        const float sc = tok[min(start + w, S - 1)];
        wts[w] = sc;
        mx = fmaxf(mx, sc);
    }
    float ssum = 0.0f;
    for (int w = 0; w <= width; ++w) {
        const float ev = __expf(wts[w] - mx);
        wts[w] = ev;
        ssum += ev;
    }
    const float inv = 1.0f / ssum;

    for (int h = t; h < H; h += 256) {
        row[h]       = (__bf16)emb[(size_t)start * H + h];
        row[H + h]   = (__bf16)emb[(size_t)end * H + h];
        float acc = 0.0f;
        for (int w = 0; w <= width; ++w)
            acc = fmaf(wts[w], emb[(size_t)min(start + w, S - 1) * H + h], acc);
        row[2 * H + h] = (__bf16)(acc * inv);
    }
}

// ---------------------------------------------------------------------------
// K4: W1T[n][k] = bf16(ms_W1[k][n])  (tiled transpose, 16x16 blocks)
// ---------------------------------------------------------------------------
__global__ void k_w1t(const float* __restrict__ W1, __bf16* __restrict__ W1T) {
    __shared__ float tile[16][17];
    const int tx = threadIdx.x, ty = threadIdx.y;
    const int k = blockIdx.x * 16 + ty;   // row of W1 (input feature)
    const int nn = blockIdx.y * 16 + tx;  // col of W1 (output neuron)
    tile[ty][tx] = W1[(size_t)k * K3H + nn];
    __syncthreads();
    W1T[(size_t)(blockIdx.y * 16 + ty) * K3H + blockIdx.x * 16 + tx] =
        (__bf16)tile[tx][ty];
}

// ---------------------------------------------------------------------------
// K5: fused mention-score GEMM, block tile 64x128, K=2304.
//     A (span_vec) staged per-block into LDS in 32-wide K chunks, double
//     buffered with GLOBAL_LOAD_ASYNC_TO_LDS_B128 (ASYNCcnt), rows padded to
//     80B for conflict-free 16-row ds_load_b128 fragment reads.
//     Each wave owns 16 columns and 4 m-subtiles: one B fragment feeds 4
//     independent v_wmma_f32_16x16x32_bf16 chains (good XDL pipelining).
//     Epilogue: relu(C + b1[col]) * w2[col], half-wave shfl_xor reduction,
//     atomicAdd into out (seeded with ms_b2).
// ---------------------------------------------------------------------------
#define AROW_E 40                 // bf16 elems per LDS row (80 B, 16B-aligned)
#define ABUF_E (64 * AROW_E)      // one K-chunk buffer: 64 rows

__global__ void __launch_bounds__(256) k_mention_gemm(
    const __bf16* __restrict__ A,  const __bf16* __restrict__ Bt,
    const float* __restrict__ b1,  const float* __restrict__ w2,
    float* __restrict__ out) {
    __shared__ __align__(16) __bf16 smemA[2 * ABUF_E];   // 10 KB

    const int lane = threadIdx.x & 31;
    const int wave = threadIdx.x >> 5;
    const int l15  = lane & 15;
    const int hi   = lane >> 4;           // 0: lanes 0-15, 1: lanes 16-31
    const int mbase = blockIdx.x * 64;
    const int ncol  = blockIdx.y * 128 + wave * 16 + l15;

    // staging map: thread t copies 16B: row = t/4, seg = t%4
    const int srow = threadIdx.x >> 2;
    const int sseg = threadIdx.x & 3;
    const __bf16* gstage = A + (size_t)(mbase + srow) * K3H + sseg * 8;
    __bf16* lstage = smemA + srow * AROW_E + sseg * 8;

    // B fragment (16-bit B 32x16 ISA layout): lane N=l15,
    //   halves hold K = hi*8 + {0..7} and hi*8 + {16..23}
    const __bf16* pb = Bt + (size_t)ncol * K3H + hi * 8;

    v8f acc[4] = {};

    // prologue: chunks 0 and 1 in flight
    stage_b128(gstage,      lstage);
    stage_b128(gstage + 32, lstage + ABUF_E);

    for (int kc = 0; kc < KSTEPS; ++kc) {
        if (kc < KSTEPS - 1) wait_async_le1();  // oldest chunk (kc) landed
        else                 wait_async_le0();
        __syncthreads();

        FragB16 fb;
        fb.q[0] = *reinterpret_cast<const uint4*>(pb);
        fb.q[1] = *reinterpret_cast<const uint4*>(pb + 16);
        if (kc + 1 < KSTEPS) __builtin_prefetch(pb + 32, 0, 1);  // global_prefetch_b8
        pb += 32;

        const __bf16* lb = smemA + (kc & 1) * ABUF_E;
#pragma unroll
        for (int mt = 0; mt < 4; ++mt) {
            // A fragment (16-bit A 16x32 ISA layout): lane M=l15,
            //   halves hold K = hi*8 + {0..7} and hi*8 + {16..23}
            const __bf16* ra = lb + (mt * 16 + l15) * AROW_E + hi * 8;
            FragB16 fa;
            fa.q[0] = *reinterpret_cast<const uint4*>(ra);
            fa.q[1] = *reinterpret_cast<const uint4*>(ra + 16);
            acc[mt] = __builtin_amdgcn_wmma_f32_16x16x32_bf16(
                          false, fa.v, false, fb.v, (short)0, acc[mt],
                          false, false);
        }
        __syncthreads();  // all waves done reading buffer (kc & 1)

        if (kc + 2 < KSTEPS)
            stage_b128(gstage + (size_t)(kc + 2) * 32,
                       lstage + (kc & 1) * ABUF_E);
    }

    // Fused epilogue: relu + scale by w2[col], reduce 16 cols per half-wave
    const float b1v = b1[ncol];
    const float w2v = w2[ncol];
#pragma unroll
    for (int mt = 0; mt < 4; ++mt) {
        float vals[8];
#pragma unroll
        for (int r = 0; r < 8; ++r) {
            float v = acc[mt][r] + b1v;
            v = v > 0.0f ? v : 0.0f;
            v *= w2v;
            v += __shfl_xor(v, 1, 32);
            v += __shfl_xor(v, 2, 32);
            v += __shfl_xor(v, 4, 32);
            v += __shfl_xor(v, 8, 32);
            vals[r] = v;
        }
        if (l15 == 0) {
            const int mb = mbase + mt * 16 + (hi ? 8 : 0);  // C: VGPR r -> M=r(+8)
#pragma unroll
            for (int r = 0; r < 8; ++r) {
                const int m = mb + r;
                if (m < NSPAN) atomicAdd(out + m, vals[r]);
            }
        }
    }
}

// ---------------------------------------------------------------------------
// Launch
// ---------------------------------------------------------------------------
extern "C" void kernel_launch(void* const* d_in, const int* in_sizes, int n_in,
                              void* d_out, int out_size, void* d_ws, size_t ws_size,
                              hipStream_t stream) {
    const float* hidden = (const float*)d_in[0];
    const float* gcnW   = (const float*)d_in[1];
    const float* gcnb   = (const float*)d_in[2];
    const float* attn_w = (const float*)d_in[3];
    const float* attn_b = (const float*)d_in[4];
    const float* msW1   = (const float*)d_in[5];
    const float* msb1   = (const float*)d_in[6];
    const float* msW2   = (const float*)d_in[7];
    const float* msb2   = (const float*)d_in[8];
    const int*   esrc   = (const int*)d_in[9];
    const int*   etgt   = (const int*)d_in[10];
    const int*   elab   = (const int*)d_in[11];
    float* out = (float*)d_out;

    char* ws = (char*)d_ws;
    float*  gcn_acc = (float*)(ws + WS_GCN);
    float*  emb     = (float*)(ws + WS_EMB);
    float*  tok     = (float*)(ws + WS_TOK);
    __bf16* spanv   = (__bf16*)(ws + WS_SPAN);
    __bf16* w1t     = (__bf16*)(ws + WS_W1T);

    k_init<<<(S * H + 255) / 256, 256, 0, stream>>>(gcn_acc, out, msb2);
    k_w1t<<<dim3(K3H / 16, K3H / 16), dim3(16, 16), 0, stream>>>(msW1, w1t);
    k_gcn_edge<<<E, 256, 0, stream>>>(hidden, gcnW, gcnb, esrc, etgt, elab, gcn_acc);
    k_emb_tok<<<S, 256, 0, stream>>>(hidden, gcn_acc, attn_w, attn_b, emb, tok);
    k_span<<<NPAD, 256, 0, stream>>>(emb, tok, spanv);
    k_mention_gemm<<<dim3(NPAD / 64, K3H / 128), 256, 0, stream>>>(
        spanv, w1t, msb1, msW2, out);
}
